// ProbAttention_5317169512517
// MI455X (gfx1250) — compile-verified, matrix operand order checked by reading, not run on hardware
//
#include <hip/hip_runtime.h>
#include <hip/hip_bf16.h>
#include <math.h>

typedef __attribute__((ext_vector_type(16))) _Float16 v16h;
typedef __attribute__((ext_vector_type(8)))  float    v8f;
typedef __attribute__((ext_vector_type(4)))  float    v4f;   // native vector for NT stores

#define B_  2
#define H_  8
#define L_  2048
#define E_  64
#define S_  40
#define U_  40
#define BH_ (B_*H_)
#define INV_LK (1.0f/2048.0f)
#define SCALE 0.125f   // 1/sqrt(64)

// ---------------- V mean per (b,h) ----------------
__global__ void vmean_kernel(const float* __restrict__ values, float* __restrict__ vmean) {
    int bh = blockIdx.x, e = threadIdx.x;
    int b = bh >> 3, h = bh & 7;
    float acc = 0.f;
    const float* p = values + ((size_t)b*L_*H_ + h)*E_ + e;
    for (int l = 0; l < L_; ++l) acc += p[(size_t)l*H_*E_];
    vmean[bh*E_ + e] = acc * INV_LK;
}

// ---------------- context = broadcast vmean (B,L,H,E) ----------------
__global__ void ctx_fill_kernel(const float* __restrict__ vmean, float* __restrict__ ctx) {
    int idx = blockIdx.x*blockDim.x + threadIdx.x;   // E=64,H=8,L=2048 powers of 2
    int e = idx & 63, h = (idx >> 6) & 7, b = idx >> 20;
    ctx[idx] = vmean[(b*H_ + h)*E_ + e];
}

// ---------------- attns = 1/L everywhere (non-temporal float4 stream) ----------------
// 268 MB > 192 MB L2: NT store policy keeps the constant bulk from thrashing L2;
// the 40 hot rows are overwritten later via the normal RT path.
__global__ void attns_fill_kernel(v4f* __restrict__ a4, long n4) {
    long i = (long)blockIdx.x*blockDim.x + threadIdx.x;
    long stride = (long)gridDim.x*blockDim.x;
    v4f v = {INV_LK, INV_LK, INV_LK, INV_LK};
    for (; i < n4; i += stride) __builtin_nontemporal_store(v, &a4[i]);
}

// ---------------- sampled sparsity measure M (one wave32 per (b,h,l)) ----------------
__global__ void msample_kernel(const float* __restrict__ q, const float* __restrict__ k,
                               const long long* __restrict__ idxs, float* __restrict__ M) {
    int wave = blockIdx.x*(blockDim.x >> 5) + (threadIdx.x >> 5);
    int lane = threadIdx.x & 31;
    int bh = wave >> 11, l = wave & (L_-1);
    int b = bh >> 3, h = bh & 7;
    const float2* qp = (const float2*)(q + (((size_t)b*L_ + l)*H_ + h)*E_);
    float2 q2 = qp[lane];
    float mx = -INFINITY, sm = 0.f;
    for (int s = 0; s < S_; ++s) {
        int kr = (int)idxs[(size_t)l*S_ + s];
        const float2* kp = (const float2*)(k + (((size_t)b*L_ + kr)*H_ + h)*E_);
        float2 k2 = kp[lane];
        float p = q2.x*k2.x + q2.y*k2.y;
        #pragma unroll
        for (int off = 16; off > 0; off >>= 1) p += __shfl_xor(p, off, 32);
        mx = fmaxf(mx, p); sm += p;
    }
    if (lane == 0) M[wave] = mx - sm*INV_LK;
}

// ---------------- iterative top-40 per (b,h), tie-break lowest index ----------------
__global__ void topk_kernel(const float* __restrict__ M, int* __restrict__ top) {
    __shared__ float smem[L_];
    __shared__ float rv[256];
    __shared__ int   ri[256];
    int bh = blockIdx.x, tid = threadIdx.x;
    for (int i = tid; i < L_; i += 256) smem[i] = M[bh*L_ + i];
    __syncthreads();
    for (int u = 0; u < U_; ++u) {
        float v = -INFINITY; int id = L_;
        for (int i = tid; i < L_; i += 256) {
            float x = smem[i];
            if (x > v || (x == v && i < id)) { v = x; id = i; }
        }
        rv[tid] = v; ri[tid] = id; __syncthreads();
        for (int s = 128; s > 0; s >>= 1) {
            if (tid < s) {
                if (rv[tid+s] > rv[tid] || (rv[tid+s] == rv[tid] && ri[tid+s] < ri[tid])) {
                    rv[tid] = rv[tid+s]; ri[tid] = ri[tid+s];
                }
            }
            __syncthreads();
        }
        if (tid == 0) { top[bh*U_ + u] = ri[0]; smem[ri[0]] = -INFINITY; }
        __syncthreads();
    }
}

// ---------------- scores = 0.125 * Q_top @ K^T via v_wmma_f32_16x16x32_f16 ----------------
// grid = BH * 3 u-tiles * 16 k-groups, block = 32 (one wave). Writes scaled scores
// directly into the selected attns rows (softmaxed in place by the next kernel).
__global__ void scores_wmma_kernel(const float* __restrict__ q, const float* __restrict__ k,
                                   const int* __restrict__ top, float* __restrict__ attns) {
    int blk = blockIdx.x;
    int kg = blk & 15, ut = (blk >> 4) % 3, bh = blk / 48;
    int b = bh >> 3, h = bh & 7;
    int lane = threadIdx.x;
    int m = lane & 15, hsel = lane >> 4;
    int u0 = ut*16, uA = u0 + m;
    int rowA = top[bh*U_ + (uA < U_ ? uA : 0)];
    const float* qrow = q + (((size_t)b*L_ + rowA)*H_ + h)*E_;
    v16h a0, a1;
    #pragma unroll
    for (int j = 0; j < 16; ++j) {
        int km = ((j >> 3) << 4) + (hsel << 3) + (j & 7);   // ISA 16-bit A layout
        a0[j] = (_Float16)qrow[km];
        a1[j] = (_Float16)qrow[32 + km];
    }
    int n = m;
    // This wave's 8 output rows (r + 8*hsel + u0) are either all < 40 or all >= 40.
    bool doStore = (u0 + (hsel << 3)) < U_;
    // Per-row destination indices (uniform over the k-tile loop).
    int rowO[8];
    #pragma unroll
    for (int r = 0; r < 8; ++r) {
        int u = u0 + r + (hsel << 3);
        rowO[r] = top[bh*U_ + (u < U_ ? u : 0)];
    }
    for (int kt = 0; kt < 8; ++kt) {
        int k0 = ((kg << 3) + kt) << 4;
        const float* krow = k + (((size_t)b*L_ + (k0 + n))*H_ + h)*E_;
        if (kt < 7) {
            // prefetch next k-tile rows (lowers to global_prefetch_b8)
            __builtin_prefetch(krow + (size_t)16*H_*E_, 0, 3);
        }
        v16h b0, b1;
        #pragma unroll
        for (int j = 0; j < 16; ++j) {
            int km = ((j >> 3) << 4) + (hsel << 3) + (j & 7);
            b0[j] = (_Float16)krow[km];
            b1[j] = (_Float16)krow[32 + km];
        }
        v8f c = {};
        c = __builtin_amdgcn_wmma_f32_16x16x32_f16(false, a0, false, b0, (short)0, c, false, false);
        c = __builtin_amdgcn_wmma_f32_16x16x32_f16(false, a1, false, b1, (short)0, c, false, false);
        if (doStore) {
            #pragma unroll
            for (int r = 0; r < 8; ++r)
                attns[((size_t)(bh*L_ + rowO[r]))*L_ + k0 + n] = c[r]*SCALE;
        }
    }
}

// ---------------- in-place softmax over each selected attns row ----------------
__global__ void softmax_kernel(const int* __restrict__ top, float* __restrict__ attns) {
    __shared__ float red[256];
    int blk = blockIdx.x, tid = threadIdx.x;
    int bh = blk / U_, u = blk % U_;
    int row = top[bh*U_ + u];
    float* p = attns + ((size_t)(bh*L_ + row))*L_;
    float x[8]; float mx = -INFINITY;
    #pragma unroll
    for (int j = 0; j < 8; ++j) { x[j] = p[tid + j*256]; mx = fmaxf(mx, x[j]); }
    red[tid] = mx; __syncthreads();
    for (int s = 128; s > 0; s >>= 1) { if (tid < s) red[tid] = fmaxf(red[tid], red[tid+s]); __syncthreads(); }
    mx = red[0]; __syncthreads();
    float sm = 0.f;
    #pragma unroll
    for (int j = 0; j < 8; ++j) { x[j] = __expf(x[j] - mx); sm += x[j]; }
    red[tid] = sm; __syncthreads();
    for (int s = 128; s > 0; s >>= 1) { if (tid < s) red[tid] += red[tid+s]; __syncthreads(); }
    float inv = 1.0f / red[0];
    #pragma unroll
    for (int j = 0; j < 8; ++j) p[tid + j*256] = x[j]*inv;
}

// ---------------- ctx_upd = attn @ V via WMMA, scattered into context ----------------
// grid = BH * 3 u-tiles * 4 e-tiles, block = 32. 64 chained WMMAs per wave (K=2048).
__global__ void ctx_wmma_kernel(const float* __restrict__ attns, const float* __restrict__ v,
                                const int* __restrict__ top, float* __restrict__ ctx) {
    int blk = blockIdx.x;
    int et = blk & 3, ut = (blk >> 2) % 3, bh = blk / 12;
    int b = bh >> 3, h = bh & 7;
    int lane = threadIdx.x;
    int m = lane & 15, hsel = lane >> 4;
    int u0 = ut*16, uA = u0 + m;
    int rowA = top[bh*U_ + (uA < U_ ? uA : 0)];
    const float* arow = attns + ((size_t)(bh*L_ + rowA))*L_;
    int n = m, e = et*16 + n;
    v8f c = {};
    for (int kc = 0; kc < L_; kc += 32) {
        v16h a, bm;
        #pragma unroll
        for (int j = 0; j < 16; ++j) {
            int km = ((j >> 3) << 4) + (hsel << 3) + (j & 7);
            a[j]  = (_Float16)arow[kc + km];
            bm[j] = (_Float16)v[(((size_t)b*L_ + (kc + km))*H_ + h)*E_ + e];
        }
        c = __builtin_amdgcn_wmma_f32_16x16x32_f16(false, a, false, bm, (short)0, c, false, false);
    }
    bool doStore = (u0 + (hsel << 3)) < U_;
    if (doStore) {
        #pragma unroll
        for (int r = 0; r < 8; ++r) {
            int u = u0 + r + (hsel << 3);
            int rowO = top[bh*U_ + (u < U_ ? u : 0)];
            ctx[(((size_t)b*L_ + rowO)*H_ + h)*E_ + et*16 + n] = c[r];
        }
    }
}

extern "C" void kernel_launch(void* const* d_in, const int* in_sizes, int n_in,
                              void* d_out, int out_size, void* d_ws, size_t ws_size,
                              hipStream_t stream) {
    const float*     q    = (const float*)d_in[0];
    const float*     k    = (const float*)d_in[1];
    const float*     v    = (const float*)d_in[2];
    const long long* idxs = (const long long*)d_in[3];   // int64 per reference

    float* ctx   = (float*)d_out;                        // (B,L,H,E) = 2,097,152
    float* attns = ctx + (size_t)B_*L_*H_*E_;            // (B,H,L,L) = 67,108,864

    float* ws_M     = (float*)d_ws;                      // BH*L
    float* ws_vmean = ws_M + BH_*L_;                     // BH*E
    int*   ws_top   = (int*)(ws_vmean + BH_*E_);         // BH*U

    vmean_kernel     <<<BH_, E_, 0, stream>>>(v, ws_vmean);
    ctx_fill_kernel  <<<(B_*L_*H_*E_)/256, 256, 0, stream>>>(ws_vmean, ctx);
    long n4 = (long)BH_*L_*L_/4;
    attns_fill_kernel<<<4096, 256, 0, stream>>>((v4f*)attns, n4);
    msample_kernel   <<<BH_*L_/8, 256, 0, stream>>>(q, k, idxs, ws_M);
    topk_kernel      <<<BH_, 256, 0, stream>>>(ws_M, ws_top);
    scores_wmma_kernel<<<BH_*3*16, 32, 0, stream>>>(q, k, ws_top, attns);
    softmax_kernel   <<<BH_*U_, 256, 0, stream>>>(ws_top, attns);
    ctx_wmma_kernel  <<<BH_*3*4, 32, 0, stream>>>(attns, v, ws_top, ctx);
}
